// LongNetViT_49435073577509
// MI455X (gfx1250) — compile-verified, matrix operand order checked by reading, not run on hardware
//
#include <hip/hip_runtime.h>
#include <hip/hip_bf16.h>
#include <math.h>

typedef __attribute__((ext_vector_type(16))) _Float16 v16h;
typedef __attribute__((ext_vector_type(8)))  _Float16 h8;
typedef __attribute__((ext_vector_type(8)))  float    v8f;

#define D_MODEL 256
#define L_SEQ   4096
#define NHEAD   8
#define HDIM    32
#define KT      128

// ------------------------------------------------------------------
// conversion kernels
// ------------------------------------------------------------------
__global__ void k_f32_to_f16_pad(const float* __restrict__ in, _Float16* __restrict__ out,
                                 long nvalid, long ntotal) {
  long i = (long)blockIdx.x * blockDim.x + threadIdx.x;
  if (i < ntotal) out[i] = (i < nvalid) ? (_Float16)in[i] : (_Float16)0.0f;
}

// in: K x N row-major f32  ->  out: N x K row-major f16
__global__ void k_transpose_f16(const float* __restrict__ in, _Float16* __restrict__ out,
                                int K, int N) {
  long i = (long)blockIdx.x * blockDim.x + threadIdx.x;
  long tot = (long)K * N;
  if (i < tot) {
    int n = (int)(i / K);
    int k = (int)(i % K);
    out[i] = (_Float16)in[(long)k * N + n];
  }
}

// ------------------------------------------------------------------
// WMMA GEMM: C[M,N] = A[M,K] (f16, row-major) * Bt[N,K]^T (f16) + bias
// optional residual add (f32), optional exact GELU; f32 and/or f16 output.
// A-fragment per ISA 16-bit A 16x32 layout: lane m(0..15)/half holds
// K = {8*half .. 8*half+7} and {16+8*half .. 16+8*half+7}. B loaded from the
// pre-transposed weight with the identical pattern (B-operand is col-striped).
// ------------------------------------------------------------------
__device__ inline v16h frag_load(const _Float16* __restrict__ base, int ld) {
  int lane = threadIdx.x & 31;
  int m = lane & 15, half = lane >> 4;
  const h8* p0 = (const h8*)(base + (long)m * ld + 8 * half);
  const h8* p1 = (const h8*)(base + (long)m * ld + 16 + 8 * half);
  h8 lo = *p0, hi = *p1;
  v16h f;
#pragma unroll
  for (int i = 0; i < 8; ++i) { f[i] = lo[i]; f[i + 8] = hi[i]; }
  return f;
}

__device__ inline float gelu_exact(float x) {
  return 0.5f * x * (1.0f + erff(x * 0.7071067811865475f));
}

__global__ void __launch_bounds__(256)
k_wmma_gemm(const _Float16* __restrict__ A, const _Float16* __restrict__ Bt,
            const float* __restrict__ bias, const float* __restrict__ res,
            float* __restrict__ outF, _Float16* __restrict__ outH,
            int M, int N, int K, int do_gelu) {
  int wave = threadIdx.x >> 5;          // 8 waves -> block tile 128 x 64
  int lane = threadIdx.x & 31;
  int row0 = blockIdx.x * 128 + wave * 16;
  int col0 = blockIdx.y * 64;
  if (row0 >= M) return;
  v8f acc0 = {}, acc1 = {}, acc2 = {}, acc3 = {};
  const _Float16* a_ptr = A  + (long)row0 * K;
  const _Float16* b0    = Bt + (long)(col0 +  0) * K;
  const _Float16* b1    = Bt + (long)(col0 + 16) * K;
  const _Float16* b2    = Bt + (long)(col0 + 32) * K;
  const _Float16* b3    = Bt + (long)(col0 + 48) * K;
  for (int k = 0; k < K; k += 32) {
    v16h a = frag_load(a_ptr + k, K);
    v16h f0 = frag_load(b0 + k, K);
    acc0 = __builtin_amdgcn_wmma_f32_16x16x32_f16(false, a, false, f0, (short)0, acc0, false, false);
    v16h f1 = frag_load(b1 + k, K);
    acc1 = __builtin_amdgcn_wmma_f32_16x16x32_f16(false, a, false, f1, (short)0, acc1, false, false);
    v16h f2 = frag_load(b2 + k, K);
    acc2 = __builtin_amdgcn_wmma_f32_16x16x32_f16(false, a, false, f2, (short)0, acc2, false, false);
    v16h f3 = frag_load(b3 + k, K);
    acc3 = __builtin_amdgcn_wmma_f32_16x16x32_f16(false, a, false, f3, (short)0, acc3, false, false);
  }
  int half = lane >> 4, nn = lane & 15;
  v8f accs[4] = {acc0, acc1, acc2, acc3};
#pragma unroll
  for (int t = 0; t < 4; ++t) {
#pragma unroll
    for (int j = 0; j < 8; ++j) {
      int r = row0 + j + 8 * half;       // C/D layout: VGPR j -> M=j (+8 for hi half)
      int c = col0 + t * 16 + nn;        // lane -> N
      float v = accs[t][j] + bias[c];
      if (res) v += res[(long)r * N + c];
      if (do_gelu) v = gelu_exact(v);
      if (outF) outF[(long)r * N + c] = v;
      if (outH) outH[(long)r * N + c] = (_Float16)v;
    }
  }
}

// ------------------------------------------------------------------
// h assembly: row 0 = cls + pos_embed[0]; rows 1..4095 = patch-GEMM + pos gather
// ------------------------------------------------------------------
__global__ void k_assemble_h(const float* __restrict__ pe_gemm, const float* __restrict__ coords,
                             const float* __restrict__ cls_token, const float* __restrict__ pos_embed,
                             float* __restrict__ h) {
  long i = (long)blockIdx.x * blockDim.x + threadIdx.x;
  if (i >= (long)L_SEQ * D_MODEL) return;
  int l = (int)(i >> 8);
  int c = (int)(i & 255);
  float v;
  if (l == 0) {
    v = cls_token[c] + pos_embed[c];
  } else {
    int t = l - 1;
    int gx = (int)floorf(coords[t * 2 + 0] * (1.0f / 256.0f));
    int gy = (int)floorf(coords[t * 2 + 1] * (1.0f / 256.0f));
    int pos = gx * 512 + gy + 1;
    v = pe_gemm[(long)t * D_MODEL + c] + pos_embed[(long)pos * D_MODEL + c];
  }
  h[i] = v;
}

// ------------------------------------------------------------------
// row LayerNorm (D=256, one block per row), f16 output for WMMA consumers
// ------------------------------------------------------------------
__global__ void __launch_bounds__(256)
k_layernorm_f16(const float* __restrict__ x, const float* __restrict__ w,
                const float* __restrict__ b, _Float16* __restrict__ out) {
  int row = blockIdx.x, tid = threadIdx.x;
  __shared__ float sred[256];
  float v = x[(long)row * D_MODEL + tid];
  sred[tid] = v; __syncthreads();
  for (int s = 128; s > 0; s >>= 1) { if (tid < s) sred[tid] += sred[tid + s]; __syncthreads(); }
  float mu = sred[0] * (1.0f / 256.0f); __syncthreads();
  float d = v - mu;
  sred[tid] = d * d; __syncthreads();
  for (int s = 128; s > 0; s >>= 1) { if (tid < s) sred[tid] += sred[tid + s]; __syncthreads(); }
  float var = sred[0] * (1.0f / 256.0f);
  out[(long)row * D_MODEL + tid] = (_Float16)(d * rsqrtf(var + 1e-5f) * w[tid] + b[tid]);
}

// ------------------------------------------------------------------
// gfx1250 async global->LDS copy of one 16-byte chunk per lane (ASYNCcnt path).
// VDST VGPR carries the LDS byte offset.
// ------------------------------------------------------------------
__device__ inline void async_copy16(const float* gptr, void* lptr) {
  unsigned lds_off = (unsigned)(size_t)lptr;  // low 32 bits of flat LDS addr = LDS offset
  asm volatile("global_load_async_to_lds_b128 %0, %1, off"
               :: "v"(lds_off), "v"(gptr) : "memory");
}

// ------------------------------------------------------------------
// dilated attention, one branch per launch. Block = 256 queries of one
// (segment, head). K/V tiles DMA'd to double-buffered LDS with
// global_load_async_to_lds_b128, prefetching tile n+1 while computing tile n.
// Online softmax per thread. All keys valid (L=4096 => pad=0 everywhere).
// qkv layout: (L, 768) with q|k|v, head h at offset h*32.
// ------------------------------------------------------------------
__global__ void __launch_bounds__(256)
k_dilated_attn(const float* __restrict__ qkv, float* __restrict__ obuf,
               float* __restrict__ lsebuf, int we, int r, int m, int off) {
  int seg = blockIdx.y;
  int hd  = blockIdx.z;
  int jq  = blockIdx.x * 256 + threadIdx.x;
  int tid = threadIdx.x;
  __shared__ float Ks[2][KT][HDIM];
  __shared__ float Vs[2][KT][HDIM];
  int base = seg * we + (hd & (r - 1));     // per-head strided offset (r is pow2)
  int lq = base + r * jq;
  float q[HDIM];
#pragma unroll
  for (int d = 0; d < HDIM; ++d) q[d] = qkv[(long)lq * 768 + hd * HDIM + d];
  float mx = -1e30f, den = 0.0f;
  float o[HDIM];
#pragma unroll
  for (int d = 0; d < HDIM; ++d) o[d] = 0.0f;
  const float scale = 0.17677669529663687f; // 1/sqrt(32)

  // issue one K/V tile (128 rows x 32 f32) as 16B async chunks: 8 chunks/row,
  // 1024 chunks per tensor, 4 per thread.
  auto issue_tile = [&](int buf, int kt) {
#pragma unroll
    for (int i = 0; i < 4; ++i) {
      int chunk = i * 256 + tid;
      int kr  = chunk >> 3;
      int sub = chunk & 7;
      int lk = base + r * (kt + kr);
      const float* gk = qkv + (long)lk * 768 + 256 + hd * HDIM + sub * 4;
      const float* gv = qkv + (long)lk * 768 + 512 + hd * HDIM + sub * 4;
      async_copy16(gk, &Ks[buf][kr][sub * 4]);
      async_copy16(gv, &Vs[buf][kr][sub * 4]);
    }
  };

  issue_tile(0, 0);
  for (int kt = 0; kt < m; kt += KT) {
    int cur = (kt / KT) & 1;
    asm volatile("s_wait_asynccnt 0x0" ::: "memory");
    __syncthreads();                         // tile `cur` complete; buffer cur^1 free
    if (kt + KT < m) issue_tile(cur ^ 1, kt + KT);
    for (int kk = 0; kk < KT; ++kk) {
      float s = 0.0f;
#pragma unroll
      for (int d = 0; d < HDIM; ++d) s += q[d] * Ks[cur][kk][d];
      s *= scale;
      float nm = fmaxf(mx, s);
      float corr = __expf(mx - nm);
      float p = __expf(s - nm);
      den = den * corr + p;
#pragma unroll
      for (int d = 0; d < HDIM; ++d) o[d] = o[d] * corr + p * Vs[cur][kk][d];
      mx = nm;
    }
  }
  long idx = off + (long)seg * m + jq;
  float inv = 1.0f / den;
#pragma unroll
  for (int d = 0; d < HDIM; ++d) obuf[(idx * NHEAD + hd) * HDIM + d] = o[d] * inv;
  lsebuf[idx * NHEAD + hd] = mx + __logf(den);
}

// ------------------------------------------------------------------
// LSE-weighted branch combine. Coverage of (l, h) by branch b:
//   (l mod we_b) mod r_b == h mod r_b
// ------------------------------------------------------------------
__global__ void __launch_bounds__(256)
k_combine(const float* __restrict__ obuf, const float* __restrict__ lsebuf,
          _Float16* __restrict__ ao16) {
  int l  = blockIdx.x;
  int hd = threadIdx.x >> 5;
  int d  = threadIdx.x & 31;
  const int WEv[5]  = {1024, 4096, 4096, 4096, 4096};
  const int RRv[5]  = {1, 2, 4, 8, 16};
  const int MMv[5]  = {1024, 2048, 1024, 512, 256};
  const int OFFv[5] = {0, 4096, 6144, 7168, 7680};
  float lse[5]; long idxv[5]; bool cov[5];
  float mx = -1e30f;
#pragma unroll
  for (int b = 0; b < 5; ++b) {
    int we = WEv[b], r = RRv[b];
    int seg = l / we, pis = l % we;
    cov[b] = ((pis & (r - 1)) == (hd & (r - 1)));
    lse[b] = -1e30f; idxv[b] = 0;
    if (cov[b]) {
      long idx = OFFv[b] + (long)seg * MMv[b] + (pis / r);
      idxv[b] = idx;
      lse[b] = lsebuf[idx * NHEAD + hd];
      mx = fmaxf(mx, lse[b]);
    }
  }
  float wsum = 0.0f, osum = 0.0f;
#pragma unroll
  for (int b = 0; b < 5; ++b) {
    if (cov[b]) {
      float w = __expf(lse[b] - mx);
      wsum += w;
      osum += w * obuf[(idxv[b] * NHEAD + hd) * HDIM + d];
    }
  }
  ao16[(long)l * D_MODEL + hd * HDIM + d] = (_Float16)(osum / wsum);
}

// ------------------------------------------------------------------
// final: double LayerNorm on cls row -> out[256]
// ------------------------------------------------------------------
__global__ void __launch_bounds__(256)
k_final_ln(const float* __restrict__ h, const float* __restrict__ ew, const float* __restrict__ eb,
           const float* __restrict__ nw, const float* __restrict__ nb, float* __restrict__ out) {
  int tid = threadIdx.x;
  __shared__ float sred[256];
  float v = h[tid];
  sred[tid] = v; __syncthreads();
  for (int s = 128; s > 0; s >>= 1) { if (tid < s) sred[tid] += sred[tid + s]; __syncthreads(); }
  float mu = sred[0] * (1.0f / 256.0f); __syncthreads();
  float d0 = v - mu;
  sred[tid] = d0 * d0; __syncthreads();
  for (int s = 128; s > 0; s >>= 1) { if (tid < s) sred[tid] += sred[tid + s]; __syncthreads(); }
  float var = sred[0] * (1.0f / 256.0f); __syncthreads();
  float y = d0 * rsqrtf(var + 1e-5f) * ew[tid] + eb[tid];
  sred[tid] = y; __syncthreads();
  for (int s = 128; s > 0; s >>= 1) { if (tid < s) sred[tid] += sred[tid + s]; __syncthreads(); }
  float mu2 = sred[0] * (1.0f / 256.0f); __syncthreads();
  float d1 = y - mu2;
  sred[tid] = d1 * d1; __syncthreads();
  for (int s = 128; s > 0; s >>= 1) { if (tid < s) sred[tid] += sred[tid + s]; __syncthreads(); }
  float var2 = sred[0] * (1.0f / 256.0f);
  out[tid] = d1 * rsqrtf(var2 + 1e-5f) * nw[tid] + nb[tid];
}

// ------------------------------------------------------------------
extern "C" void kernel_launch(void* const* d_in, const int* in_sizes, int n_in,
                              void* d_out, int out_size, void* d_ws, size_t ws_size,
                              hipStream_t stream) {
  (void)in_sizes; (void)n_in; (void)out_size; (void)ws_size;
  const float* x       = (const float*)d_in[0];
  const float* coords  = (const float*)d_in[1];
  const float* patch_w = (const float*)d_in[2];
  const float* patch_b = (const float*)d_in[3];
  const float* cls_tok = (const float*)d_in[4];
  const float* pos_emb = (const float*)d_in[5];
  const float* ln1_w   = (const float*)d_in[6];
  const float* ln1_b   = (const float*)d_in[7];
  const float* wqkv    = (const float*)d_in[8];
  const float* bqkv    = (const float*)d_in[9];
  const float* wo      = (const float*)d_in[10];
  const float* bo      = (const float*)d_in[11];
  const float* ln2_w   = (const float*)d_in[12];
  const float* ln2_b   = (const float*)d_in[13];
  const float* w1      = (const float*)d_in[14];
  const float* b1      = (const float*)d_in[15];
  const float* w2      = (const float*)d_in[16];
  const float* b2      = (const float*)d_in[17];
  const float* encln_w = (const float*)d_in[18];
  const float* encln_b = (const float*)d_in[19];
  const float* norm_w  = (const float*)d_in[20];
  const float* norm_b  = (const float*)d_in[21];
  float* out = (float*)d_out;

  char* ws = (char*)d_ws;
  size_t off = 0;
  auto alloc = [&](size_t bytes) -> void* {
    off = (off + 255) & ~(size_t)255;
    void* p = ws + off; off += bytes; return p;
  };
  _Float16* x16    = (_Float16*)alloc((size_t)4096 * 1536 * 2);
  _Float16* wptT   = (_Float16*)alloc((size_t)256 * 1536 * 2);
  _Float16* wqkvT  = (_Float16*)alloc((size_t)4 * 768 * 256 * 2);
  _Float16* woT    = (_Float16*)alloc((size_t)4 * 256 * 256 * 2);
  _Float16* w1T    = (_Float16*)alloc((size_t)4 * 1024 * 256 * 2);
  _Float16* w2T    = (_Float16*)alloc((size_t)4 * 256 * 1024 * 2);
  float*    hbuf   = (float*)alloc((size_t)4096 * 256 * 4);
  float*    pe_out = (float*)alloc((size_t)4096 * 256 * 4);
  _Float16* xn16   = (_Float16*)alloc((size_t)4096 * 256 * 2);
  float*    qkv    = (float*)alloc((size_t)4096 * 768 * 4);
  float*    obuf   = (float*)alloc((size_t)7936 * 8 * 32 * 4);
  float*    lsebuf = (float*)alloc((size_t)7936 * 8 * 4);
  _Float16* ao16   = (_Float16*)alloc((size_t)4096 * 256 * 2);
  _Float16* m16    = (_Float16*)alloc((size_t)4096 * 1024 * 2);

  const int TB = 256;
  auto cdiv = [](long a, long b) { return (int)((a + b - 1) / b); };

  // one-time f16 conversions / weight transposes
  k_f32_to_f16_pad<<<cdiv((long)4096 * 1536, TB), TB, 0, stream>>>(x, x16, (long)4095 * 1536, (long)4096 * 1536);
  k_transpose_f16<<<cdiv((long)1536 * 256, TB), TB, 0, stream>>>(patch_w, wptT, 1536, 256);
  for (int l = 0; l < 4; ++l) {
    k_transpose_f16<<<cdiv((long)256 * 768, TB), TB, 0, stream>>>(wqkv + (long)l * 256 * 768, wqkvT + (long)l * 768 * 256, 256, 768);
    k_transpose_f16<<<cdiv((long)256 * 256, TB), TB, 0, stream>>>(wo + (long)l * 256 * 256, woT + (long)l * 256 * 256, 256, 256);
    k_transpose_f16<<<cdiv((long)256 * 1024, TB), TB, 0, stream>>>(w1 + (long)l * 256 * 1024, w1T + (long)l * 1024 * 256, 256, 1024);
    k_transpose_f16<<<cdiv((long)1024 * 256, TB), TB, 0, stream>>>(w2 + (long)l * 1024 * 256, w2T + (long)l * 256 * 1024, 1024, 256);
  }

  // patch embed GEMM + assemble
  dim3 g_pe(32, 4);
  k_wmma_gemm<<<g_pe, 256, 0, stream>>>(x16, wptT, patch_b, nullptr, pe_out, nullptr, 4096, 256, 1536, 0);
  k_assemble_h<<<cdiv((long)4096 * 256, TB), TB, 0, stream>>>(pe_out, coords, cls_tok, pos_emb, hbuf);

  const int WEv[5]  = {1024, 4096, 4096, 4096, 4096};
  const int RRv[5]  = {1, 2, 4, 8, 16};
  const int MMv[5]  = {1024, 2048, 1024, 512, 256};
  const int NNv[5]  = {4, 1, 1, 1, 1};
  const int OFFv[5] = {0, 4096, 6144, 7168, 7680};

  for (int l = 0; l < 4; ++l) {
    k_layernorm_f16<<<4096, 256, 0, stream>>>(hbuf, ln1_w + l * 256, ln1_b + l * 256, xn16);
    dim3 g_qkv(32, 12);
    k_wmma_gemm<<<g_qkv, 256, 0, stream>>>(xn16, wqkvT + (long)l * 768 * 256, bqkv + l * 768, nullptr, qkv, nullptr, 4096, 768, 256, 0);
    for (int b = 0; b < 5; ++b) {
      dim3 g_at(MMv[b] / 256, NNv[b], NHEAD);
      k_dilated_attn<<<g_at, 256, 0, stream>>>(qkv, obuf, lsebuf, WEv[b], RRv[b], MMv[b], OFFv[b]);
    }
    k_combine<<<4096, 256, 0, stream>>>(obuf, lsebuf, ao16);
    dim3 g_o(32, 4);
    k_wmma_gemm<<<g_o, 256, 0, stream>>>(ao16, woT + (long)l * 256 * 256, bo + l * 256, hbuf, hbuf, nullptr, 4096, 256, 256, 0);
    k_layernorm_f16<<<4096, 256, 0, stream>>>(hbuf, ln2_w + l * 256, ln2_b + l * 256, xn16);
    dim3 g_m1(32, 16);
    k_wmma_gemm<<<g_m1, 256, 0, stream>>>(xn16, w1T + (long)l * 1024 * 256, b1 + l * 1024, nullptr, nullptr, m16, 4096, 1024, 256, 1);
    dim3 g_m2(32, 4);
    k_wmma_gemm<<<g_m2, 256, 0, stream>>>(m16, w2T + (long)l * 256 * 1024, b2 + l * 256, hbuf, hbuf, nullptr, 4096, 256, 1024, 0);
  }
  k_final_ln<<<1, 256, 0, stream>>>(hbuf, encln_w, encln_b, norm_w, norm_b, out);
}
